// TFCRNN_39024072851695
// MI455X (gfx1250) — matrix operand order repaired and checked by reading, not compiled, not run on hardware
//
#include <hip/hip_runtime.h>
#include <hip/hip_bf16.h>

// Problem dims (from reference)
#define B_   32
#define T_   512
#define IN_  256
#define H_   2048
#define C_   16
#define G3_  6144   // 3*H
#define KTF  32     // teacher-forcing K padded 16 -> 32 for one WMMA

// Persistent scan config: 128 WGs, each owns H/128 = 16 h-columns.
// 4 wave32s per WG: wave = (kh<<1)|mi ; mi = M tile (batch rows), kh = K half.
#define NWG    128
#define HCHUNK 16

typedef __bf16 bf16_t;
typedef __attribute__((ext_vector_type(16))) __bf16 v16bf;
typedef __attribute__((ext_vector_type(8)))  __bf16 v8bf;
typedef __attribute__((ext_vector_type(8)))  float  v8f;

// Exact operand type of the async-LDS builtin on this toolchain (from the
// round-2 diagnostic): 16-byte int vector, global (as1) src / LDS (as3) dst.
typedef int b128v __attribute__((vector_size(16)));
typedef __attribute__((address_space(1))) b128v* gb128p;
typedef __attribute__((address_space(3))) b128v* lb128p;

#if defined(__AMDGCN__) && __has_builtin(__builtin_amdgcn_global_load_async_to_lds_b128)
#define HAVE_ASYNC_LDS 1
#else
#define HAVE_ASYNC_LDS 0
#endif

union AB16 {
    v16bf v;
    struct { v8bf lo; v8bf hi; } p;
};

__device__ __forceinline__ v8f zero8() {
    v8f z = {0.f, 0.f, 0.f, 0.f, 0.f, 0.f, 0.f, 0.f};
    return z;
}

// A-matrix fragment (16x32 bf16, M x K), source row-major [rows][ld].
// ISA layout: lane l holds row M0+(l&15); lanes<16: K = {0..7, 16..23},
// lanes>=16: K = {8..15, 24..31} (relative to k0).
__device__ __forceinline__ v16bf load_frag_a(const bf16_t* base, int ld,
                                             int m0, int k0, int lane) {
    int row = m0 + (lane & 15);
    int off = (lane >> 4) ? 8 : 0;
    const bf16_t* p = base + (size_t)row * ld + k0 + off;
    AB16 u;
    u.p.lo = *(const v8bf*)(p);        // elements 0..7
    u.p.hi = *(const v8bf*)(p + 16);   // elements 8..15
    return u.v;
}

// B-matrix fragment (32x16 bf16, K x N). Source row-major [N][ld]
// (row = output/gate column, contiguous K). Lane l holds column N0+(l&15);
// lanes<16: K = {0..7, 16..23}, lanes>=16: K = {8..15, 24..31} (relative to k0).
__device__ __forceinline__ v16bf load_frag_b(const bf16_t* base, int ld,
                                             int n0, int k0, int lane) {
    int row = n0 + (lane & 15);
    int off = (lane >> 4) ? 8 : 0;
    const bf16_t* p = base + (size_t)row * ld + k0 + off;
    AB16 u;
    u.p.lo = *(const v8bf*)(p);
    u.p.hi = *(const v8bf*)(p + 16);
    return u.v;
}

__device__ __forceinline__ v8f wmma_bf16(v16bf a, v16bf b, v8f c) {
    return __builtin_amdgcn_wmma_f32_16x16x32_bf16(
        /*neg_a=*/false, a, /*neg_b=*/false, b,
        /*c_mod=*/(short)0, c, /*reuse_a=*/false, /*reuse_b=*/false);
}

// 16-byte global -> LDS stage; async DMA path if the toolchain exposes it.
__device__ __forceinline__ void stage16(bf16_t* dst_lds, const bf16_t* src_glob) {
#if HAVE_ASYNC_LDS
    __builtin_amdgcn_global_load_async_to_lds_b128(
        (gb128p)(unsigned long long)src_glob,
        (lb128p)(unsigned int)(unsigned long long)dst_lds,
        0, 0);
#else
    *(v8bf*)dst_lds = *(const v8bf*)src_glob;
#endif
}

__device__ __forceinline__ void stage_wait() {
#if HAVE_ASYNC_LDS
#if __has_builtin(__builtin_amdgcn_s_wait_asynccnt)
    __builtin_amdgcn_s_wait_asynccnt(0);
#else
    asm volatile("s_wait_asynccnt 0" ::: "memory");
#endif
#endif
}

// ---------------------------------------------------------------------------
// Prep: fp32 -> bf16 weight/activation conversion + state zeroing.
// ---------------------------------------------------------------------------
__global__ void tfcrnn_prep_kernel(const float* __restrict__ x,
                                   const float* __restrict__ W_ih,
                                   const float* __restrict__ W_hh,
                                   const float* __restrict__ W_cls,
                                   bf16_t* __restrict__ whh,
                                   bf16_t* __restrict__ wx,
                                   bf16_t* __restrict__ wtf,
                                   bf16_t* __restrict__ wcls,
                                   bf16_t* __restrict__ xb,
                                   float* __restrict__ hf,
                                   bf16_t* __restrict__ hb,
                                   unsigned int* __restrict__ ctr,
                                   long long total) {
    const long long S0 = (long long)G3_ * H_;
    const long long S1 = S0 + (long long)G3_ * IN_;
    const long long S2 = S1 + (long long)G3_ * KTF;
    const long long S3 = S2 + (long long)C_ * H_;
    const long long S4 = S3 + (long long)T_ * B_ * IN_;
    long long i = (long long)blockIdx.x * blockDim.x + threadIdx.x;
    long long stride = (long long)gridDim.x * blockDim.x;
    for (; i < total; i += stride) {
        if (i < S0) {
            whh[i] = (bf16_t)W_hh[i];
        } else if (i < S1) {
            long long j = i - S0;
            long long r = j / IN_, c = j % IN_;
            wx[j] = (bf16_t)W_ih[r * (IN_ + C_) + c];
        } else if (i < S2) {
            long long j = i - S1;
            long long r = j / KTF, c = j % KTF;
            wtf[j] = (c < C_) ? (bf16_t)W_ih[r * (IN_ + C_) + IN_ + c]
                              : (bf16_t)0.0f;
        } else if (i < S3) {
            long long j = i - S2;
            wcls[j] = (bf16_t)W_cls[j];
        } else if (i < S4) {
            long long j = i - S3;
            long long t = j / ((long long)B_ * IN_);
            long long rem = j % ((long long)B_ * IN_);
            long long b = rem / IN_, c = rem % IN_;
            xb[j] = (bf16_t)x[(b * T_ + t) * IN_ + c];
        } else {
            long long j = i - S4;           // [0, 2*B*H)
            hf[j] = 0.0f;
            hb[j] = (bf16_t)0.0f;
            if (j == 0) *ctr = 0u;
        }
    }
}

// ---------------------------------------------------------------------------
// Persistent GRU scan. Grid = NWG x 128 threads (4 wave32s).
// This WG's bf16 weight slices live in LDS for the whole scan (CDNA5 WGP has
// 320KB LDS): W_hh slice 192KB + W_x 24KB + W_tf 3KB + W_cls 64KB (+tf, +red).
// Waves split the K dimension in half and reduce partials through LDS so all
// four SIMD32s of the WGP stay busy at 1 workgroup/WGP occupancy.
// ---------------------------------------------------------------------------
#define RIDX(mi, a, lane) ((((mi) * 4 + (a)) * 32 + (lane)) * 8)

__global__ void __launch_bounds__(128, 1)
tfcrnn_scan_kernel(const float* __restrict__ y,
                   const int* __restrict__ rv,
                   const float* __restrict__ b_ih,
                   const float* __restrict__ b_hh,
                   const float* __restrict__ b_cls,
                   const bf16_t* __restrict__ whh,
                   const bf16_t* __restrict__ wx,
                   const bf16_t* __restrict__ wtf,
                   const bf16_t* __restrict__ wcls,
                   const bf16_t* __restrict__ xb,
                   float* __restrict__ hf,
                   bf16_t* __restrict__ hb,
                   unsigned int* __restrict__ ctr,
                   float* __restrict__ out) {
    // LDS-resident weight slices (rows 0..15 = r gate, 16..31 = z, 32..47 = n)
    __shared__ __align__(16) bf16_t lds_whh[48 * H_];     // 196608 B
    __shared__ __align__(16) bf16_t lds_wx [48 * IN_];    //  24576 B
    __shared__ __align__(16) bf16_t lds_wtf[48 * KTF];    //   3072 B
    __shared__ __align__(16) bf16_t lds_wcls[C_ * H_];    //  65536 B
    __shared__ __align__(16) bf16_t lds_tf[B_ * KTF];     //   2048 B
    __shared__ __align__(32) float  lds_red[2 * 4 * 32 * 8]; // 8192 B

    const int tid  = threadIdx.x;
    const int lane = tid & 31;
    const int wave = tid >> 5;
    const int wg   = blockIdx.x;
    const int mi   = wave & 1;        // M tile: batch rows mi*16..mi*16+15
    const int kh   = wave >> 1;       // K half
    const int j0   = wg * HCHUNK;     // first h-column owned by this WG
    const int jlane = j0 + (lane & 15);
    const int hi8  = (lane >> 4) ? 8 : 0;

    // ---- one-time LDS fill of this WG's weight slices (async if available) --
    for (int i = tid; i < 48 * (H_ / 8); i += blockDim.x) {     // W_hh slice
        int r = i >> 8, c = (i & 255) << 3;
        int g = r >> 4, rr = r & 15;
        stage16(&lds_whh[r * H_ + c],
                whh + (size_t)(g * H_ + j0 + rr) * H_ + c);
    }
    for (int i = tid; i < 48 * (IN_ / 8); i += blockDim.x) {    // W_x slice
        int r = i >> 5, c = (i & 31) << 3;
        int g = r >> 4, rr = r & 15;
        stage16(&lds_wx[r * IN_ + c],
                wx + (size_t)(g * H_ + j0 + rr) * IN_ + c);
    }
    for (int i = tid; i < 48 * (KTF / 8); i += blockDim.x) {    // W_tf slice
        int r = i >> 2, c = (i & 3) << 3;
        int g = r >> 4, rr = r & 15;
        stage16(&lds_wtf[r * KTF + c],
                wtf + (size_t)(g * H_ + j0 + rr) * KTF + c);
    }
    for (int i = tid; i < C_ * (H_ / 8); i += blockDim.x) {     // W_cls (full)
        stage16(&lds_wcls[i * 8], wcls + (size_t)i * 8);
    }
    for (int i = tid; i < B_ * KTF; i += blockDim.x)            // tf(0) = 0
        lds_tf[i] = (bf16_t)0.0f;
    stage_wait();
    __syncthreads();

    // Hoisted per-lane biases.
    const float bir = b_ih[jlane],          bhr = b_hh[jlane];
    const float biz = b_ih[H_ + jlane],     bhz = b_hh[H_ + jlane];
    const float bin = b_ih[2 * H_ + jlane], bhn = b_hh[2 * H_ + jlane];
    const float bcl = b_cls[lane & 15];

    for (int t = 0; t < T_; ++t) {
        const int cur = t & 1;
        const int nxt = cur ^ 1;
        const bf16_t* hb_cur = hb + (size_t)cur * B_ * H_;
        const float*  hf_cur = hf + (size_t)cur * B_ * H_;
        bf16_t* hb_nxt = hb + (size_t)nxt * B_ * H_;
        float*  hf_nxt = hf + (size_t)nxt * B_ * H_;

        v8f ar = zero8(), az = zero8(), ani = zero8(), anh = zero8();

        // ---- recurrent GEMM (this wave's K half): A from global h, B from LDS
        const int kb = kh * (H_ / 2);
        for (int kk = 0; kk < H_ / 2; kk += 32) {
            const int k = kb + kk;
            v16bf a = load_frag_a(hb_cur, H_, mi * 16, k, lane);
            ar  = wmma_bf16(a, load_frag_b(lds_whh, H_, 0,  k, lane), ar);
            az  = wmma_bf16(a, load_frag_b(lds_whh, H_, 16, k, lane), az);
            anh = wmma_bf16(a, load_frag_b(lds_whh, H_, 32, k, lane), anh);
        }
        // ---- input GEMM (K half of 256) ----
        const bf16_t* xt = xb + (size_t)t * B_ * IN_;
        const int xkb = kh * (IN_ / 2);
        for (int kk = 0; kk < IN_ / 2; kk += 32) {
            const int k = xkb + kk;
            v16bf a = load_frag_a(xt, IN_, mi * 16, k, lane);
            ar  = wmma_bf16(a, load_frag_b(lds_wx, IN_, 0,  k, lane), ar);
            az  = wmma_bf16(a, load_frag_b(lds_wx, IN_, 16, k, lane), az);
            ani = wmma_bf16(a, load_frag_b(lds_wx, IN_, 32, k, lane), ani);
        }
        // ---- teacher-forcing GEMM: one K step, kh==0 waves only ----
        if (kh == 0) {
            v16bf a = load_frag_a(lds_tf, KTF, mi * 16, 0, lane);
            ar  = wmma_bf16(a, load_frag_b(lds_wtf, KTF, 0,  0, lane), ar);
            az  = wmma_bf16(a, load_frag_b(lds_wtf, KTF, 16, 0, lane), az);
            ani = wmma_bf16(a, load_frag_b(lds_wtf, KTF, 32, 0, lane), ani);
        }

        // ---- cross-wave K reduction through LDS ----
        if (kh == 1) {
            *(v8f*)&lds_red[RIDX(mi, 0, lane)] = ar;
            *(v8f*)&lds_red[RIDX(mi, 1, lane)] = az;
            *(v8f*)&lds_red[RIDX(mi, 2, lane)] = ani;
            *(v8f*)&lds_red[RIDX(mi, 3, lane)] = anh;
        }
        __syncthreads();

        // ---- GRU gate math + h update (kh==0 waves; 32x16 tile of h) ----
        if (kh == 0) {
            ar  += *(const v8f*)&lds_red[RIDX(mi, 0, lane)];
            az  += *(const v8f*)&lds_red[RIDX(mi, 1, lane)];
            ani += *(const v8f*)&lds_red[RIDX(mi, 2, lane)];
            anh += *(const v8f*)&lds_red[RIDX(mi, 3, lane)];
#pragma unroll
            for (int e = 0; e < 8; ++e) {
                int brow = mi * 16 + hi8 + e;
                float rr = 1.0f / (1.0f + __expf(-(ar[e] + bir + bhr)));
                float zz = 1.0f / (1.0f + __expf(-(az[e] + biz + bhz)));
                float nn = tanhf(ani[e] + bin + rr * (anh[e] + bhn));
                float hold = hf_cur[(size_t)brow * H_ + jlane];
                float hnew = (1.0f - zz) * nn + zz * hold;
                hf_nxt[(size_t)brow * H_ + jlane] = hnew;
                hb_nxt[(size_t)brow * H_ + jlane] = (bf16_t)hnew;
            }
        }

        // ---- grid-wide barrier: h[nxt] fully written across all WGs ----
        __threadfence();
        __syncthreads();
        if (tid == 0) {
            __hip_atomic_fetch_add(ctr, 1u, __ATOMIC_ACQ_REL,
                                   __HIP_MEMORY_SCOPE_AGENT);
            const unsigned int target =
                (unsigned int)(t + 1) * (unsigned int)gridDim.x;
            while (__hip_atomic_load(ctr, __ATOMIC_ACQUIRE,
                                     __HIP_MEMORY_SCOPE_AGENT) < target) {
                __builtin_amdgcn_s_sleep(2);
            }
        }
        __syncthreads();

        // ---- classifier (redundant per WG, K split over waves) ----
        v8f ac = zero8();
        for (int kk = 0; kk < H_ / 2; kk += 32) {
            const int k = kb + kk;
            v16bf a = load_frag_a(hb_nxt, H_, mi * 16, k, lane);
            ac = wmma_bf16(a, load_frag_b(lds_wcls, H_, 0, k, lane), ac);
        }
        if (kh == 1) {
            *(v8f*)&lds_red[RIDX(mi, 0, lane)] = ac;
        }
        __syncthreads();
        if (kh == 0) {
            ac += *(const v8f*)&lds_red[RIDX(mi, 0, lane)];
            // scheduled sampling prob for step t
            float p = (float)t * 0.01f;
            float d = __expf(-6.0f * p);
            float prob = fminf(0.9f, 1.0f - fminf(0.05f, 2.0f / (1.0f + d) - 1.0f));
            int c = lane & 15;
#pragma unroll
            for (int e = 0; e < 8; ++e) {
                int brow = mi * 16 + hi8 + e;
                float val = ac[e] + bcl;
                if (wg == 0)
                    out[(size_t)brow * T_ * C_ + (size_t)t * C_ + c] = val;
                float sig = (val > 0.0f) ? 1.0f : 0.0f;   // sigmoid(val) > 0.5
                bool flag = ((float)rv[t * B_ + brow] * 0.001f) < prob;
                float tfv = flag
                    ? y[(size_t)brow * T_ * C_ + (size_t)t * C_ + c] : sig;
                lds_tf[brow * KTF + c] = (bf16_t)tfv;
            }
        }
        __syncthreads();
    }
}

extern "C" void kernel_launch(void* const* d_in, const int* in_sizes, int n_in,
                              void* d_out, int out_size, void* d_ws,
                              size_t ws_size, hipStream_t stream) {
    const float* x     = (const float*)d_in[0];
    const float* y     = (const float*)d_in[1];
    const int*   rv    = (const int*)  d_in[2];
    const float* W_ih  = (const float*)d_in[3];
    const float* W_hh  = (const float*)d_in[4];
    const float* b_ih  = (const float*)d_in[5];
    const float* b_hh  = (const float*)d_in[6];
    const float* W_cls = (const float*)d_in[7];
    const float* b_cls = (const float*)d_in[8];
    float* out = (float*)d_out;

    char* ws = (char*)d_ws;
    size_t off = 0;
    auto alloc = [&](size_t bytes) -> void* {
        void* p = ws + off;
        off += bytes;
        off = (off + 255) & ~(size_t)255;
        return p;
    };
    bf16_t* whh  = (bf16_t*)alloc((size_t)G3_ * H_ * sizeof(bf16_t));
    bf16_t* wx   = (bf16_t*)alloc((size_t)G3_ * IN_ * sizeof(bf16_t));
    bf16_t* wtf  = (bf16_t*)alloc((size_t)G3_ * KTF * sizeof(bf16_t));
    bf16_t* wcls = (bf16_t*)alloc((size_t)C_ * H_ * sizeof(bf16_t));
    bf16_t* xb   = (bf16_t*)alloc((size_t)T_ * B_ * IN_ * sizeof(bf16_t));
    float*  hf   = (float*) alloc((size_t)2 * B_ * H_ * sizeof(float));
    bf16_t* hb   = (bf16_t*)alloc((size_t)2 * B_ * H_ * sizeof(bf16_t));
    unsigned int* ctr = (unsigned int*)alloc(256);
    (void)ws_size; (void)in_sizes; (void)n_in; (void)out_size;

    const long long total =
        (long long)G3_ * H_ + (long long)G3_ * IN_ + (long long)G3_ * KTF +
        (long long)C_ * H_ + (long long)T_ * B_ * IN_ + (long long)2 * B_ * H_;

    tfcrnn_prep_kernel<<<8192, 256, 0, stream>>>(
        x, W_ih, W_hh, W_cls, whh, wx, wtf, wcls, xb, hf, hb, ctr, total);

    tfcrnn_scan_kernel<<<NWG, 128, 0, stream>>>(
        y, rv, b_ih, b_hh, b_cls, whh, wx, wtf, wcls, xb, hf, hb, ctr, out);
}
